// QuantumNet_71296457114137
// MI455X (gfx1250) — matrix operand order repaired and checked by reading, not compiled, not run on hardware
//
#include <hip/hip_runtime.h>
#include <hip/hip_bf16.h>

// ---------------------------------------------------------------------------
// Fused QuantumNet kernel for gfx1250 (MI455X), single pass over HBM.
//   pre_out = X @ pre_W^T + pre_b          (WMMA f32_16x16x32_f16, N=4 used)
//   q_in    = tanh(pre_out) * pi/2
//   q_out   = 4-qubit statevector sim      (16 amps in registers / thread)
//   out     = q_out @ post_W^T + post_b
// Memory-bound: 128 MiB X read once -> ~5.8us floor @ 23.3 TB/s.
// B operand is zero-padded to 16 cols in LDS (branchless fragment build),
// with row pitch 520 halves so half-wave b128 reads are LDS-bank conflict-free.
// ---------------------------------------------------------------------------

typedef __attribute__((ext_vector_type(16))) _Float16 v16h;
typedef __attribute__((ext_vector_type(8)))  float    v8f;

#define NQ 4
#define QDEPTH 6
#define DDIM 512
#define WPITCH (DDIM + 8)    // halves; 260 words -> lane n starts at bank 4n
#define ROWS_PER_BLOCK 256   // 8 waves x 32 rows
#define THREADS 256

__device__ __forceinline__ void apply_ry(float st[16], const int str, float c, float s) {
#pragma unroll
  for (int i = 0; i < 16; ++i) {
    if ((i & str) == 0) {
      float a0 = st[i], a1 = st[i + str];
      st[i]       = c * a0 - s * a1;
      st[i + str] = s * a0 + c * a1;
    }
  }
}

__device__ __forceinline__ void apply_cnot(float st[16], const int cs, const int ts) {
#pragma unroll
  for (int i = 0; i < 16; ++i) {
    if ((i & cs) != 0 && (i & ts) == 0) {   // control=1, target=0 : swap pair
      float t = st[i]; st[i] = st[i + ts]; st[i + ts] = t;
    }
  }
}

__global__ __launch_bounds__(THREADS)
void quantumnet_fused(const float* __restrict__ X,       // (B, 512)
                      const float* __restrict__ preW,    // (4, 512)
                      const float* __restrict__ preB,    // (4,)
                      const float* __restrict__ qparams, // (24,)
                      const float* __restrict__ postW,   // (2, 4)
                      const float* __restrict__ postB,   // (2,)
                      float* __restrict__ out)           // (B, 2)
{
  __shared__ _Float16 sW[16 * WPITCH];            // pre_W^T cols 0..3, zeros 4..15
  __shared__ float    sAng[ROWS_PER_BLOCK * NQ];  // per-row q_in angles
  __shared__ float    sQc[QDEPTH * NQ], sQs[QDEPTH * NQ];

  const int tid = threadIdx.x;

  // --- one-time block setup: zero-padded f16 weights + trig of qparams ------
  for (int i = tid; i < 16 * WPITCH; i += THREADS) {
    int r = i / WPITCH, c = i - r * WPITCH;
    float v = (r < NQ && c < DDIM) ? preW[r * DDIM + c] : 0.0f;
    sW[i] = (_Float16)v;
  }
  if (tid < QDEPTH * NQ) {
    float h = qparams[tid] * 0.5f;
    sQc[tid] = cosf(h);
    sQs[tid] = sinf(h);
  }
  __syncthreads();

  const int wave = tid >> 5;       // 0..7
  const int lane = tid & 31;
  const int n    = lane & 15;      // WMMA column (qubit index when < 4) / A row
  const int hi   = lane >> 4;      // half-wave selector

  const long rowBlock = (long)blockIdx.x * ROWS_PER_BLOCK + wave * 32;
  const float* __restrict__ a0 = X + (rowBlock +      n) * (long)DDIM; // tile 0 row
  const float* __restrict__ a1 = X + (rowBlock + 16 + n) * (long)DDIM; // tile 1 row
  const _Float16* __restrict__ wrow = &sW[n * WPITCH + hi * 8];

  // --- pre-GEMM: two 16x16 accumulators, K = 512 in chunks of 32 ------------
  v8f c0 = {}; v8f c1 = {};
  for (int k = 0; k < DDIM; k += 32) {
    __builtin_prefetch(a0 + k + 128, 0, 0);
    __builtin_prefetch(a1 + k + 128, 0, 0);

    // B fragment (branchless): lane col n, halves 0-7 <-> K = k + hi*8 + h, 8-15 <-> k + 16 + hi*8 + (h-8)
    v16h bf;
    {
      const _Float16* wp = wrow + k;
#pragma unroll
      for (int h = 0; h < 16; ++h) bf[h] = wp[h < 8 ? h : h + 8];
    }

    // A fragments: lane row m = n; halves 0-7 <-> K = hi*8+j, 8-15 <-> 16+hi*8+j
    v16h af0, af1;
    {
      const float4* p = reinterpret_cast<const float4*>(a0 + k + hi * 8);
      float4 x0 = p[0], x1 = p[1];
      const float4* q = reinterpret_cast<const float4*>(a0 + k + 16 + hi * 8);
      float4 y0 = q[0], y1 = q[1];
      af0[0]=(_Float16)x0.x; af0[1]=(_Float16)x0.y; af0[2]=(_Float16)x0.z; af0[3]=(_Float16)x0.w;
      af0[4]=(_Float16)x1.x; af0[5]=(_Float16)x1.y; af0[6]=(_Float16)x1.z; af0[7]=(_Float16)x1.w;
      af0[8]=(_Float16)y0.x; af0[9]=(_Float16)y0.y; af0[10]=(_Float16)y0.z; af0[11]=(_Float16)y0.w;
      af0[12]=(_Float16)y1.x; af0[13]=(_Float16)y1.y; af0[14]=(_Float16)y1.z; af0[15]=(_Float16)y1.w;
    }
    {
      const float4* p = reinterpret_cast<const float4*>(a1 + k + hi * 8);
      float4 x0 = p[0], x1 = p[1];
      const float4* q = reinterpret_cast<const float4*>(a1 + k + 16 + hi * 8);
      float4 y0 = q[0], y1 = q[1];
      af1[0]=(_Float16)x0.x; af1[1]=(_Float16)x0.y; af1[2]=(_Float16)x0.z; af1[3]=(_Float16)x0.w;
      af1[4]=(_Float16)x1.x; af1[5]=(_Float16)x1.y; af1[6]=(_Float16)x1.z; af1[7]=(_Float16)x1.w;
      af1[8]=(_Float16)y0.x; af1[9]=(_Float16)y0.y; af1[10]=(_Float16)y0.z; af1[11]=(_Float16)y0.w;
      af1[12]=(_Float16)y1.x; af1[13]=(_Float16)y1.y; af1[14]=(_Float16)y1.z; af1[15]=(_Float16)y1.w;
    }

    c0 = __builtin_amdgcn_wmma_f32_16x16x32_f16(false, af0, false, bf, (short)0, c0, false, false);
    c1 = __builtin_amdgcn_wmma_f32_16x16x32_f16(false, af1, false, bf, (short)0, c1, false, false);
  }

  // --- bias + tanh + scale, scatter angles to per-row LDS layout -----------
  const float HALF_PI = 1.57079632679489662f;
  if (n < NQ) {
    float bias = preB[n];
#pragma unroll
    for (int r = 0; r < 8; ++r) {
      int m0 = wave * 32 + r + 8 * hi;        // local row, tile 0
      int m1 = m0 + 16;                        // local row, tile 1
      sAng[m0 * NQ + n] = tanhf(c0[r] + bias) * HALF_PI;
      sAng[m1 * NQ + n] = tanhf(c1[r] + bias) * HALF_PI;
    }
  }
  __syncthreads();

  // --- per-thread 4-qubit statevector simulation ---------------------------
  float ang[NQ];
#pragma unroll
  for (int q = 0; q < NQ; ++q) ang[q] = sAng[tid * NQ + q];

  // H layer + data RY layer on |0000> is a product state:
  //   per-wire vector = RY(theta) * H|0> = ((c-s)/sqrt2, (c+s)/sqrt2)
  const float RS = 0.70710678118654752f;
  float u0[NQ], u1[NQ];
#pragma unroll
  for (int w = 0; w < NQ; ++w) {
    float hc = cosf(ang[w] * 0.5f), hs = sinf(ang[w] * 0.5f);
    u0[w] = (hc - hs) * RS;
    u1[w] = (hc + hs) * RS;
  }
  float st[16];
#pragma unroll
  for (int i = 0; i < 16; ++i) {
    st[i] = ((i & 8) ? u1[0] : u0[0]) * ((i & 4) ? u1[1] : u0[1]) *
            ((i & 2) ? u1[2] : u0[2]) * ((i & 1) ? u1[3] : u0[3]);
  }

  // depth x (CNOT(0,1), CNOT(2,3), CNOT(1,2) + trainable RY layer)
#pragma unroll
  for (int d = 0; d < QDEPTH; ++d) {
    apply_cnot(st, 8, 4);   // c=0 (mask 8), t=1 (mask 4)
    apply_cnot(st, 2, 1);   // c=2, t=3
    apply_cnot(st, 4, 2);   // c=1, t=2
#pragma unroll
    for (int w = 0; w < NQ; ++w)
      apply_ry(st, 8 >> w, sQc[d * NQ + w], sQs[d * NQ + w]);
  }

  // <Z_w> expectations
  float z[NQ] = {0.f, 0.f, 0.f, 0.f};
#pragma unroll
  for (int i = 0; i < 16; ++i) {
    float p = st[i] * st[i];
    z[0] += (i & 8) ? -p : p;
    z[1] += (i & 4) ? -p : p;
    z[2] += (i & 2) ? -p : p;
    z[3] += (i & 1) ? -p : p;
  }

  // post projection (2x4) + bias
  float o0 = postB[0], o1 = postB[1];
#pragma unroll
  for (int w = 0; w < NQ; ++w) {
    o0 += z[w] * postW[w];
    o1 += z[w] * postW[NQ + w];
  }

  const long row = (long)blockIdx.x * ROWS_PER_BLOCK + tid;
  reinterpret_cast<float2*>(out)[row] = make_float2(o0, o1);
}

extern "C" void kernel_launch(void* const* d_in, const int* in_sizes, int n_in,
                              void* d_out, int out_size, void* d_ws, size_t ws_size,
                              hipStream_t stream) {
  const float* X     = (const float*)d_in[0];  // (B, 512)
  const float* preW  = (const float*)d_in[1];  // (4, 512)
  const float* preB  = (const float*)d_in[2];  // (4,)
  const float* qp    = (const float*)d_in[3];  // (24,)
  const float* postW = (const float*)d_in[4];  // (2, 4)
  const float* postB = (const float*)d_in[5];  // (2,)
  float* out = (float*)d_out;

  const int B = in_sizes[0] / DDIM;            // 65536
  const int blocks = B / ROWS_PER_BLOCK;       // 256 (B is a multiple of 256)

  quantumnet_fused<<<blocks, THREADS, 0, stream>>>(X, preW, preB, qp, postW, postB, out);
}
